// Attn_33028298506245
// MI455X (gfx1250) — compile-verified, matrix operand order checked by reading, not run on hardware
//
#include <hip/hip_runtime.h>

typedef __bf16 bf16;
typedef __attribute__((ext_vector_type(8))) __bf16 bf16x8;
typedef __attribute__((ext_vector_type(16))) __bf16 bf16x16;
typedef __attribute__((ext_vector_type(8))) float f32x8;
typedef int v4i __attribute__((ext_vector_type(4)));

constexpr int T = 2048, C = 1024, H = 16, HD = 96, VD = 96, SKEEP = 512;

__device__ __forceinline__ f32x8 zero8() {
  f32x8 z = {0.f, 0.f, 0.f, 0.f, 0.f, 0.f, 0.f, 0.f};
  return z;
}

__device__ __forceinline__ f32x8 wmma_bf16(bf16x16 a, bf16x16 b, f32x8 c) {
  return __builtin_amdgcn_wmma_f32_16x16x32_bf16(false, a, false, b, (short)0, c,
                                                 false, false);
}

// A 16x32 bf16 fragment is, per lane, two contiguous 16B chunks:
//   chunk0 = elements [k0 + 8h .. +7], chunk1 = elements [k0 + 16 + 8h .. +7]
// and fragment slots f[0..7] / f[8..15] are exactly those chunks in order
// (CDNA5 ISA VGPR layout tables, CDNA5 ISA §7.12.2). Load them as explicit
// b128 vector loads and concatenate.
__device__ __forceinline__ bf16x16 frag_from_chunks(const bf16* p0) {
  bf16x8 lo = *(const bf16x8*)p0;
  bf16x8 hi = *(const bf16x8*)(p0 + 16);
  return __builtin_shufflevector(lo, hi, 0, 1, 2, 3, 4, 5, 6, 7, 8, 9, 10, 11,
                                 12, 13, 14, 15);
}

// A fragment: 16x32 bf16 tile from row-major A[M x K]
__device__ __forceinline__ bf16x16 load_a(const bf16* A, int lda, int row0, int k0) {
  int lane = threadIdx.x & 31;
  int m = row0 + (lane & 15), h = lane >> 4;
  return frag_from_chunks(A + (size_t)m * lda + k0 + 8 * h);
}

// B fragment from a TRANSPOSED operand: logical B[K x N] stored as Bt[N x K].
__device__ __forceinline__ bf16x16 load_bT(const bf16* Bt, int ldk, int col0, int k0) {
  int lane = threadIdx.x & 31;
  int n = col0 + (lane & 15), h = lane >> 4;
  return frag_from_chunks(Bt + (size_t)n * ldk + k0 + 8 * h);
}

__device__ __forceinline__ void store_c(float* __restrict__ Cm, int N, int tm,
                                        int tn, f32x8 acc) {
  int lane = threadIdx.x & 31;
  int n = tn + (lane & 15), h = lane >> 4;
#pragma unroll
  for (int r = 0; r < 8; ++r)
    Cm[(tm + r + 8 * h) * N + n] = acc[r];
}

// ---------------- WMMA GEMM: C[M,N] = A[M,K] @ Bt[N,K]^T, 32x32 tile/wave ----
__global__ __launch_bounds__(32) void gemm_bf16_k(const bf16* __restrict__ A,
                                                  const bf16* __restrict__ Bt,
                                                  float* __restrict__ Cm, int M,
                                                  int N, int K) {
  (void)M;
  int tn = blockIdx.x << 5, tm = blockIdx.y << 5;
  int lane = threadIdx.x & 31;
  f32x8 acc00 = zero8(), acc01 = zero8(), acc10 = zero8(), acc11 = zero8();
  for (int k0 = 0; k0 < K; k0 += 32) {
    if (k0 + 32 < K) {
      __builtin_prefetch(A + (size_t)(tm + lane) * K + k0 + 32, 0, 1);
      __builtin_prefetch(Bt + (size_t)(tn + lane) * K + k0 + 32, 0, 1);
    }
    bf16x16 a0 = load_a(A, K, tm, k0);
    bf16x16 a1 = load_a(A, K, tm + 16, k0);
    bf16x16 b0 = load_bT(Bt, K, tn, k0);
    bf16x16 b1 = load_bT(Bt, K, tn + 16, k0);
    acc00 = wmma_bf16(a0, b0, acc00);
    acc01 = wmma_bf16(a0, b1, acc01);
    acc10 = wmma_bf16(a1, b0, acc10);
    acc11 = wmma_bf16(a1, b1, acc11);
  }
  store_c(Cm, N, tm, tn, acc00);
  store_c(Cm, N, tm, tn + 16, acc01);
  store_c(Cm, N, tm + 16, tn, acc10);
  store_c(Cm, N, tm + 16, tn + 16, acc11);
}

// ---------------- elementwise f32 -> bf16 ------------------------------------
__global__ void cvt_k(const float* __restrict__ s, bf16* __restrict__ d, int n) {
  int i = blockIdx.x * 256 + threadIdx.x;
  if (i < n) d[i] = (bf16)s[i];
}

// ---------------- transpose+convert: src[K x N] f32 -> dst[N x K] bf16 -------
__global__ void tr_cvt_k(const float* __restrict__ src, bf16* __restrict__ dst,
                         int K, int N) {
  int i = blockIdx.x * 256 + threadIdx.x;
  if (i >= K * N) return;
  int k = i % K, n = i / K;
  dst[i] = (bf16)src[k * N + n];
}

// ---------------- RMSNorm: wave per row --------------------------------------
__global__ void rms_k(const float* __restrict__ in, const float* __restrict__ scale,
                      bf16* __restrict__ out, int rows, int d) {
  int wave = threadIdx.x >> 5, lane = threadIdx.x & 31;
  int row = blockIdx.x * 8 + wave;
  if (row >= rows) return;
  float ss = 0.f;
  for (int j = lane; j < d; j += 32) {
    float v = in[row * d + j];
    ss += v * v;
  }
#pragma unroll
  for (int off = 16; off; off >>= 1) ss += __shfl_xor(ss, off, 32);
  float r = rsqrtf(ss / (float)d + 1e-6f);
  for (int j = lane; j < d; j += 32)
    out[row * d + j] = (bf16)(in[row * d + j] * r * scale[j]);
}

// ---------------- pack (separate nope/rope mats) + RoPE, [h][row][96] --------
__global__ void pack_split_k(const float* __restrict__ nope,
                             const float* __restrict__ rope, bf16* __restrict__ out,
                             int rows) {
  int i = blockIdx.x * 256 + threadIdx.x;
  int total = H * rows * HD;
  if (i >= total) return;
  int d = i % HD;
  int t = (i / HD) % rows;
  int h = i / (HD * rows);
  float val;
  if (d < 32) {
    val = nope[t * (H * 32) + h * 32 + d];
  } else {
    int j = d - 32, jj = j & 31;
    float re = rope[t * (H * 64) + h * 64 + jj];
    float im = rope[t * (H * 64) + h * 64 + 32 + jj];
    float inv = __expf(-(float)(2 * jj) * (9.210340371976184f / 64.f));
    float ang = (float)t * inv;
    float cs = __cosf(ang), sn = __sinf(ang);
    val = (j < 32) ? (re * cs - im * sn) : (re * sn + im * cs);
  }
  out[(h * rows + t) * HD + d] = (bf16)val;
}

// ---------------- pack fused [rows x 1536] K (rope on last 64 of each head) --
__global__ void pack_fused_k(const float* __restrict__ in, bf16* __restrict__ out,
                             int rows) {
  int i = blockIdx.x * 256 + threadIdx.x;
  int total = H * rows * HD;
  if (i >= total) return;
  int d = i % HD;
  int t = (i / HD) % rows;
  int h = i / (HD * rows);
  const float* rowp = in + t * 1536 + h * 96;
  float val;
  if (d < 32) {
    val = rowp[d];
  } else {
    int j = d - 32, jj = j & 31;
    float re = rowp[32 + jj];
    float im = rowp[64 + jj];
    float inv = __expf(-(float)(2 * jj) * (9.210340371976184f / 64.f));
    float ang = (float)t * inv;
    float cs = __cosf(ang), sn = __sinf(ang);
    val = (j < 32) ? (re * cs - im * sn) : (re * sn + im * cs);
  }
  out[(h * rows + t) * HD + d] = (bf16)val;
}

// ---------------- pack V transposed: [rows x 1536] -> [h][d=96][rows] --------
__global__ void pack_v_k(const float* __restrict__ in, bf16* __restrict__ out,
                         int rows) {
  int i = blockIdx.x * 256 + threadIdx.x;
  int total = H * rows * VD;
  if (i >= total) return;
  int t = i % rows;            // contiguous along rows for coalesced writes
  int d = (i / rows) % VD;
  int h = i / (rows * VD);
  out[(h * VD + d) * rows + t] = (bf16)in[t * 1536 + h * 96 + d];
}

// ---------------- importance scores: wave per row ----------------------------
__global__ void scores_k(const float* __restrict__ x, const float* __restrict__ wimp,
                         const float* __restrict__ bimp, float* __restrict__ scores) {
  int wave = threadIdx.x >> 5, lane = threadIdx.x & 31;
  int row = blockIdx.x * 8 + wave;
  if (row >= T) return;
  float s = 0.f;
  for (int c = lane; c < C; c += 32) s += x[row * C + c] * wimp[c];
#pragma unroll
  for (int off = 16; off; off >>= 1) s += __shfl_xor(s, off, 32);
  if (!lane) scores[row] = s + bimp[0];
}

// ---------------- top-512 of 2048 via bitonic sort, then sort indices --------
__global__ void topk_k(const float* __restrict__ scores, int* __restrict__ idx_out) {
  __shared__ float ks[T];
  __shared__ int ki[T];
  __shared__ int si[SKEEP];
  int tid = threadIdx.x;
  for (int i = tid; i < T; i += 256) {
    ks[i] = scores[i];
    ki[i] = i;
  }
  __syncthreads();
  // descending sort, ties -> smaller index first
  for (int k = 2; k <= T; k <<= 1) {
    for (int j = k >> 1; j > 0; j >>= 1) {
      for (int i = tid; i < T; i += 256) {
        int ixj = i ^ j;
        if (ixj > i) {
          bool up = ((i & k) == 0);
          float s1 = ks[i], s2 = ks[ixj];
          int i1 = ki[i], i2 = ki[ixj];
          bool less1 = (s1 < s2) || (s1 == s2 && i1 > i2);
          if (up == less1) {
            ks[i] = s2; ks[ixj] = s1;
            ki[i] = i2; ki[ixj] = i1;
          }
        }
      }
      __syncthreads();
    }
  }
  for (int i = tid; i < SKEEP; i += 256) si[i] = ki[i];
  __syncthreads();
  // ascending sort of surviving indices
  for (int k = 2; k <= SKEEP; k <<= 1) {
    for (int j = k >> 1; j > 0; j >>= 1) {
      for (int i = tid; i < SKEEP; i += 256) {
        int ixj = i ^ j;
        if (ixj > i) {
          bool up = ((i & k) == 0);
          int a = si[i], b = si[ixj];
          if (up == (a > b)) { si[i] = b; si[ixj] = a; }
        }
      }
      __syncthreads();
    }
  }
  for (int i = tid; i < SKEEP; i += 256) idx_out[i] = si[i];
}

// ---------------- gather selected tokens via async global->LDS staging -------
__global__ __launch_bounds__(128) void gather_k(const bf16* __restrict__ xb,
                                                const int* __restrict__ idx,
                                                bf16* __restrict__ selb) {
  __shared__ __align__(16) bf16 buf[C];
  int s = blockIdx.x;
  int t = idx[s];
  int e = threadIdx.x * 8; // 8 bf16 = 16 bytes per lane; 128 lanes cover C=1024
  const bf16* g = xb + t * C + e;
#if __has_builtin(__builtin_amdgcn_global_load_async_to_lds_b128)
  __builtin_amdgcn_global_load_async_to_lds_b128(
      (__attribute__((address_space(1))) v4i*)g,
      (__attribute__((address_space(3))) v4i*)(buf + e), 0, 0);
#else
  {
    unsigned lds_off =
        (unsigned)(__UINTPTR_TYPE__)(__attribute__((address_space(3))) bf16*)(buf + e);
    asm volatile("global_load_async_to_lds_b128 %0, %1, off" ::"v"(lds_off),
                 "v"(g)
                 : "memory");
  }
#endif
#if __has_builtin(__builtin_amdgcn_s_wait_asynccnt)
  __builtin_amdgcn_s_wait_asynccnt(0);
#else
  asm volatile("s_wait_asynccnt 0" ::: "memory");
#endif
  __syncthreads();
  const uint4 v = *reinterpret_cast<const uint4*>(buf + e);
  *reinterpret_cast<uint4*>(selb + s * C + e) = v;
}

// ---------------- gate: logits then softmax ----------------------------------
__global__ void gate_logits_k(const float* __restrict__ x, const float* __restrict__ wg,
                              const float* __restrict__ bg, float* __restrict__ logits) {
  __shared__ float red[256];
  int j = blockIdx.x;
  float s = 0.f;
  for (int l = threadIdx.x; l < T * C; l += 256) {
    int c = l & (C - 1);
    s += x[l] * wg[c * 3 + j];
  }
  red[threadIdx.x] = s;
  __syncthreads();
  for (int st = 128; st; st >>= 1) {
    if (threadIdx.x < st) red[threadIdx.x] += red[threadIdx.x + st];
    __syncthreads();
  }
  if (!threadIdx.x) logits[j] = red[0] / (float)T + bg[j];
}

__global__ void gate_softmax_k(const float* __restrict__ logits,
                               float* __restrict__ gate) {
  if (threadIdx.x == 0) {
    float m = fmaxf(logits[0], fmaxf(logits[1], logits[2]));
    float e0 = __expf(logits[0] - m), e1 = __expf(logits[1] - m),
          e2 = __expf(logits[2] - m);
    float s = e0 + e1 + e2;
    gate[0] = e0 / s;
    gate[1] = e1 / s;
    gate[2] = e2 / s;
  }
}

// ---------------- flash attention (WMMA, online softmax) ---------------------
// One wave per (head, 16-query tile). K is [h][t][96] row-major (contiguous
// B^T fragments); V is pre-transposed [h][d][S] so P@V B fragments are also
// contiguous b128 loads. P is bounced through LDS into A-fragment layout.
__global__ __launch_bounds__(32) void flash_k(const bf16* __restrict__ Q,
                                              const bf16* __restrict__ Kt,
                                              const bf16* __restrict__ Vt,
                                              float* __restrict__ O,
                                              const float* __restrict__ gate,
                                              int gidx, int S, int causal,
                                              int accum) {
  __shared__ __align__(16) bf16 plds[16 * 40];
  const int head = blockIdx.y;
  const int q0 = blockIdx.x << 4;
  const bf16* Qh = Q + (size_t)head * T * HD;
  const bf16* Kh = Kt + (size_t)head * S * HD;
  const bf16* Vh = Vt + (size_t)head * VD * S; // [d][S]
  float* Oh = O + (size_t)head * T * VD;
  const int lane = threadIdx.x & 31;
  const int n = lane & 15, hh = lane >> 4;
  const float scl = 0.1020620726159657f; // 1/sqrt(96)

  bf16x16 aq0 = load_a(Qh, HD, q0, 0);
  bf16x16 aq1 = load_a(Qh, HD, q0, 32);
  bf16x16 aq2 = load_a(Qh, HD, q0, 64);

  f32x8 oacc[6];
#pragma unroll
  for (int v = 0; v < 6; ++v) oacc[v] = zero8();
  float mi[8], li[8];
#pragma unroll
  for (int r = 0; r < 8; ++r) { mi[r] = -1e30f; li[r] = 0.f; }

  const int kend = causal ? (q0 + 16) : S;
  for (int key0 = 0; key0 < kend; key0 += 32) {
    if (key0 + 64 <= S) {
      __builtin_prefetch(Kh + (size_t)(key0 + 32 + lane) * HD, 0, 1);
      __builtin_prefetch(Vh + (size_t)(lane * 3) * S + key0 + 32, 0, 1);
    }
    f32x8 s0 = zero8(), s1 = zero8();
    {
      bf16x16 b0 = load_bT(Kh, HD, key0, 0);
      bf16x16 b1 = load_bT(Kh, HD, key0 + 16, 0);
      s0 = wmma_bf16(aq0, b0, s0);
      s1 = wmma_bf16(aq0, b1, s1);
      b0 = load_bT(Kh, HD, key0, 32);
      b1 = load_bT(Kh, HD, key0 + 16, 32);
      s0 = wmma_bf16(aq1, b0, s0);
      s1 = wmma_bf16(aq1, b1, s1);
      b0 = load_bT(Kh, HD, key0, 64);
      b1 = load_bT(Kh, HD, key0 + 16, 64);
      s0 = wmma_bf16(aq2, b0, s0);
      s1 = wmma_bf16(aq2, b1, s1);
    }
#pragma unroll
    for (int r = 0; r < 8; ++r) {
      int qi = q0 + r + 8 * hh;
      float e0 = s0[r] * scl;
      float e1 = s1[r] * scl;
      if (causal) {
        if (key0 + n > qi) e0 = -1e30f;
        if (key0 + 16 + n > qi) e1 = -1e30f;
      }
      float mx = fmaxf(e0, e1);
#pragma unroll
      for (int off = 1; off < 16; off <<= 1) mx = fmaxf(mx, __shfl_xor(mx, off, 32));
      float mN = fmaxf(mi[r], mx);
      float p0 = __expf(e0 - mN), p1 = __expf(e1 - mN);
      float rs = p0 + p1;
#pragma unroll
      for (int off = 1; off < 16; off <<= 1) rs += __shfl_xor(rs, off, 32);
      float alpha = __expf(mi[r] - mN);
      li[r] = li[r] * alpha + rs;
      mi[r] = mN;
#pragma unroll
      for (int v = 0; v < 6; ++v) oacc[v][r] *= alpha;
      plds[(r + 8 * hh) * 40 + n] = (bf16)p0;
      plds[(r + 8 * hh) * 40 + 16 + n] = (bf16)p1;
    }
    __syncthreads();
    // P fragment: A-layout chunks live at plds[n*40 + 8*hh] and +16 -> 2x
    // ds_load_b128 per lane.
    bf16x16 pf = frag_from_chunks(plds + n * 40 + 8 * hh);
#pragma unroll
    for (int v = 0; v < 6; ++v) {
      bf16x16 bv = load_bT(Vh, S, v * 16, key0); // V^T fragment, b128 loads
      oacc[v] = wmma_bf16(pf, bv, oacc[v]);
    }
    __syncthreads();
  }
  float g = gate[gidx];
#pragma unroll
  for (int r = 0; r < 8; ++r) {
    float inv = li[r] > 0.f ? (g / li[r]) : 0.f;
    int row = q0 + r + 8 * hh;
#pragma unroll
    for (int v = 0; v < 6; ++v) {
      int o = row * VD + v * 16 + n;
      float val = oacc[v][r] * inv;
      if (accum)
        Oh[o] += val;
      else
        Oh[o] = val;
    }
  }
}

// ---------------- concat heads -> [T, 1536] bf16 -----------------------------
__global__ void concat_k(const float* __restrict__ o_f, bf16* __restrict__ ob) {
  int i = blockIdx.x * 256 + threadIdx.x;
  int total = H * T * VD;
  if (i >= total) return;
  int d = i % VD;
  int t = (i / VD) % T;
  int h = i / (VD * T);
  ob[t * 1536 + h * 96 + d] = (bf16)o_f[(h * T + t) * VD + d];
}

// =============================================================================
extern "C" void kernel_launch(void* const* d_in, const int* in_sizes, int n_in,
                              void* d_out, int out_size, void* d_ws, size_t ws_size,
                              hipStream_t stream) {
  (void)in_sizes; (void)n_in; (void)out_size; (void)ws_size;
  const float* x = (const float*)d_in[0];
  const float* w_cq = (const float*)d_in[1];
  const float* q_scale = (const float*)d_in[2];
  const float* w_dq_n = (const float*)d_in[3];
  const float* w_dq_r = (const float*)d_in[4];
  const float* w_ckv = (const float*)d_in[5];
  const float* kv_scale = (const float*)d_in[6];
  const float* w_dk_n = (const float*)d_in[7];
  const float* w_dv = (const float*)d_in[8];
  const float* w_k_r = (const float*)d_in[9];
  const float* w_imp = (const float*)d_in[10];
  const float* b_imp = (const float*)d_in[11];
  const float* w_sel_k = (const float*)d_in[12];
  const float* w_sel_v = (const float*)d_in[13];
  const float* w_win_k = (const float*)d_in[14];
  const float* w_win_v = (const float*)d_in[15];
  const float* w_gate = (const float*)d_in[16];
  const float* b_gate = (const float*)d_in[17];
  const float* w_proj = (const float*)d_in[18];
  float* out = (float*)d_out;

  char* p = (char*)d_ws;
  auto alloc = [&](size_t bytes) -> char* {
    char* r = p;
    p += (bytes + 255) & ~(size_t)255;
    return r;
  };
  bf16* xb = (bf16*)alloc((size_t)T * C * 2);
  // all weight matrices stored TRANSPOSED ([N x K]) in bf16
  bf16* wcq = (bf16*)alloc((size_t)C * 96 * 2);
  bf16* wdqn = (bf16*)alloc((size_t)96 * 512 * 2);
  bf16* wdqr = (bf16*)alloc((size_t)96 * 1024 * 2);
  bf16* wckv = (bf16*)alloc((size_t)C * 32 * 2);
  bf16* wdkn = (bf16*)alloc((size_t)32 * 512 * 2);
  bf16* wdv = (bf16*)alloc((size_t)32 * 1536 * 2);
  bf16* wkr = (bf16*)alloc((size_t)32 * 1024 * 2);
  bf16* wsk = (bf16*)alloc((size_t)C * 1536 * 2);
  bf16* wsv = (bf16*)alloc((size_t)C * 1536 * 2);
  bf16* wwk = (bf16*)alloc((size_t)C * 1536 * 2);
  bf16* wwv = (bf16*)alloc((size_t)C * 1536 * 2);
  bf16* wpj = (bf16*)alloc((size_t)1536 * 1024 * 2);
  float* tmp1 = (float*)alloc((size_t)T * 1536 * 4);
  float* tmp2 = (float*)alloc((size_t)T * 1024 * 4);
  bf16* qnb = (bf16*)alloc((size_t)T * 96 * 2);
  bf16* kvnb = (bf16*)alloc((size_t)T * 32 * 2);
  bf16* qh = (bf16*)alloc((size_t)H * T * 96 * 2);
  bf16* k1h = (bf16*)alloc((size_t)H * T * 96 * 2);
  bf16* v1h = (bf16*)alloc((size_t)H * T * 96 * 2);
  bf16* kwh = (bf16*)alloc((size_t)H * T * 96 * 2);
  bf16* vwh = (bf16*)alloc((size_t)H * T * 96 * 2);
  bf16* ksh = (bf16*)alloc((size_t)H * SKEEP * 96 * 2);
  bf16* vsh = (bf16*)alloc((size_t)H * SKEEP * 96 * 2);
  bf16* selb = (bf16*)alloc((size_t)SKEEP * C * 2);
  float* scores = (float*)alloc((size_t)T * 4);
  int* idxb = (int*)alloc((size_t)SKEEP * 4);
  float* logits = (float*)alloc(16);
  float* gate = (float*)alloc(16);
  float* o_f = (float*)alloc((size_t)H * T * 96 * 4);
  bf16* ob = (bf16*)alloc((size_t)T * 1536 * 2);

  cvt_k<<<(T * C + 255) / 256, 256, 0, stream>>>(x, xb, T * C);
  auto trcvt = [&](const float* s, bf16* d, int Kd, int Nd) {
    tr_cvt_k<<<(Kd * Nd + 255) / 256, 256, 0, stream>>>(s, d, Kd, Nd);
  };
  trcvt(w_cq, wcq, C, 96);
  trcvt(w_dq_n, wdqn, 96, 512);
  trcvt(w_dq_r, wdqr, 96, 1024);
  trcvt(w_ckv, wckv, C, 32);
  trcvt(w_dk_n, wdkn, 32, 512);
  trcvt(w_dv, wdv, 32, 1536);
  trcvt(w_k_r, wkr, 32, 1024);
  trcvt(w_sel_k, wsk, C, 1536);
  trcvt(w_sel_v, wsv, C, 1536);
  trcvt(w_win_k, wwk, C, 1536);
  trcvt(w_win_v, wwv, C, 1536);
  trcvt(w_proj, wpj, 1536, 1024);

  // compressed q / kv + RMSNorm
  gemm_bf16_k<<<dim3(96 / 32, T / 32), 32, 0, stream>>>(xb, wcq, tmp1, T, 96, C);
  rms_k<<<T / 8, 256, 0, stream>>>(tmp1, q_scale, qnb, T, 96);
  gemm_bf16_k<<<dim3(32 / 32, T / 32), 32, 0, stream>>>(xb, wckv, tmp1, T, 32, C);
  rms_k<<<T / 8, 256, 0, stream>>>(tmp1, kv_scale, kvnb, T, 32);

  // Q decompression + RoPE pack
  gemm_bf16_k<<<dim3(512 / 32, T / 32), 32, 0, stream>>>(qnb, wdqn, tmp1, T, 512, 96);
  gemm_bf16_k<<<dim3(1024 / 32, T / 32), 32, 0, stream>>>(qnb, wdqr, tmp2, T, 1024, 96);
  pack_split_k<<<(H * T * 96 + 255) / 256, 256, 0, stream>>>(tmp1, tmp2, qh, T);

  // branch 1: compressed KV
  gemm_bf16_k<<<dim3(512 / 32, T / 32), 32, 0, stream>>>(kvnb, wdkn, tmp1, T, 512, 32);
  gemm_bf16_k<<<dim3(1024 / 32, T / 32), 32, 0, stream>>>(kvnb, wkr, tmp2, T, 1024, 32);
  pack_split_k<<<(H * T * 96 + 255) / 256, 256, 0, stream>>>(tmp1, tmp2, k1h, T);
  gemm_bf16_k<<<dim3(1536 / 32, T / 32), 32, 0, stream>>>(kvnb, wdv, tmp1, T, 1536, 32);
  pack_v_k<<<(H * T * 96 + 255) / 256, 256, 0, stream>>>(tmp1, v1h, T);

  // branch 2: top-k selected tokens
  scores_k<<<T / 8, 256, 0, stream>>>(x, w_imp, b_imp, scores);
  topk_k<<<1, 256, 0, stream>>>(scores, idxb);
  gather_k<<<SKEEP, 128, 0, stream>>>(xb, idxb, selb);
  gemm_bf16_k<<<dim3(1536 / 32, SKEEP / 32), 32, 0, stream>>>(selb, wsk, tmp1, SKEEP, 1536, C);
  pack_fused_k<<<(H * SKEEP * 96 + 255) / 256, 256, 0, stream>>>(tmp1, ksh, SKEEP);
  gemm_bf16_k<<<dim3(1536 / 32, SKEEP / 32), 32, 0, stream>>>(selb, wsv, tmp1, SKEEP, 1536, C);
  pack_v_k<<<(H * SKEEP * 96 + 255) / 256, 256, 0, stream>>>(tmp1, vsh, SKEEP);

  // branch 3: window (full causal)
  gemm_bf16_k<<<dim3(1536 / 32, T / 32), 32, 0, stream>>>(xb, wwk, tmp1, T, 1536, C);
  pack_fused_k<<<(H * T * 96 + 255) / 256, 256, 0, stream>>>(tmp1, kwh, T);
  gemm_bf16_k<<<dim3(1536 / 32, T / 32), 32, 0, stream>>>(xb, wwv, tmp1, T, 1536, C);
  pack_v_k<<<(H * T * 96 + 255) / 256, 256, 0, stream>>>(tmp1, vwh, T);

  // gate
  gate_logits_k<<<3, 256, 0, stream>>>(x, w_gate, b_gate, logits);
  gate_softmax_k<<<1, 32, 0, stream>>>(logits, gate);

  // three gated attentions accumulated into o_f
  flash_k<<<dim3(T / 16, H), 32, 0, stream>>>(qh, k1h, v1h, o_f, gate, 0, T, 1, 0);
  flash_k<<<dim3(T / 16, H), 32, 0, stream>>>(qh, ksh, vsh, o_f, gate, 1, SKEEP, 0, 1);
  flash_k<<<dim3(T / 16, H), 32, 0, stream>>>(qh, kwh, vwh, o_f, gate, 2, T, 1, 1);

  // concat heads + output projection
  concat_k<<<(H * T * 96 + 255) / 256, 256, 0, stream>>>(o_f, ob);
  gemm_bf16_k<<<dim3(C / 32, T / 32), 32, 0, stream>>>(ob, wpj, out, T, C, 1536);
}
